// LocalSelfAttention_9706626089585
// MI455X (gfx1250) — matrix operand order repaired and checked by reading, not compiled, not run on hardware
//
#include <hip/hip_runtime.h>

// Longformer sliding-window attention for gfx1250 (MI455X).
// Compute-bound (~42 GFLOP over ~60MB, fits in 192MB L2) -> use
// v_wmma_f32_16x16x32_f16 (f16 in, f32 accumulate) throughout.

typedef _Float16 h16;
typedef __attribute__((ext_vector_type(16))) _Float16 v16h;
typedef __attribute__((ext_vector_type(8)))  float    v8f;

#define SEQL   4096
#define BATCH  2
#define HEADS  12
#define HDIM   64
#define EMB    768
#define MROWS  (SEQL*BATCH)      // 8192
#define KPAD   256               // low-side pad for key rows
#define KROWS  4640              // 4096 + 256 lo + 288 hi
#define MASKNEG (-3.402823466e38f)
#define NEGINF  (-__builtin_inff())

// ---- WMMA fragment helpers (contiguous-k order per lane half, wave32) -------------------
// A (16x32 f16): lane L holds row M=L%16; element e -> K = 16*(L/16)+e
__device__ inline v16h load_a_frag(const h16* row, int k0, int g) {
  v16h r;
  const h16* p0 = row + k0 + 16*g;       // e=0..7  -> 8 contiguous halves
  const h16* p1 = row + k0 + 16*g + 8;   // e=8..15 -> 8 contiguous halves
#pragma unroll
  for (int i = 0; i < 8; ++i) { r[i] = p0[i]; r[8+i] = p1[i]; }
  return r;
}
// B (32x16 f16): lane L holds col N=L%16; element e -> K = 16*(L/16)+e
// (callers arrange memory so the 16 halves are contiguous)
__device__ inline v16h load_b_frag(const h16* p) {
  v16h r;
#pragma unroll
  for (int i = 0; i < 16; ++i) r[i] = p[i];
  return r;
}
__device__ inline v8f wmma_f16(v16h a, v16h b, v8f c) {
  return __builtin_amdgcn_wmma_f32_16x16x32_f16(false, a, false, b, (short)0, c,
                                                false, false);
}

// ---- Kernel 1: fp32 -> f16 activations ------------------------------------
__global__ void conv_val(const float* __restrict__ in, h16* __restrict__ out) {
  int i = blockIdx.x * blockDim.x + threadIdx.x;
  out[i] = (h16)in[i];
}

// ---- Kernel 2: fp32 -> f16 weights, swizzled to B-fragment order ----------
// B[k][n] = Worig[n][k].  Storage: block (n16,k16) of 16x16 holds, for each
// column c, the 16 k-halves contiguous:  wsw[((n16*48+k16)*256) + c*16 + e].
__global__ void conv_w(const float* __restrict__ Wq, const float* __restrict__ Wk,
                       const float* __restrict__ Wv, h16* __restrict__ wsw) {
  int i = blockIdx.x * blockDim.x + threadIdx.x;
  const int per = EMB * EMB;
  int z = i / per, r = i % per;
  int blk = r >> 8, wi = r & 255;
  int cc = wi >> 4, e = wi & 15;
  int n16 = blk / 48, k16 = blk % 48;
  const float* src = (z == 0) ? Wq : (z == 1) ? Wk : Wv;
  float scale = (z == 0) ? 0.125f : 1.0f;   // fold 1/sqrt(64) into Wq
  wsw[(size_t)z * per + r] = (h16)(src[(n16*16 + cc)*EMB + k16*16 + e] * scale);
}

// ---- Kernel 3: QKV projection GEMM (8192x768 @ 768x768) -------------------
// grid (64, 12, 3), 256 threads = 8 waves; wave -> 16x64 output tile.
__global__ __launch_bounds__(256) void qkv_gemm(
    const h16* __restrict__ valh, const h16* __restrict__ wsw,
    const float* __restrict__ bq, const float* __restrict__ bk,
    const float* __restrict__ bv,
    h16* __restrict__ qbuf, h16* __restrict__ kbuf, h16* __restrict__ vtbuf) {
  const int lane = threadIdx.x & 31;
  const int wave = threadIdx.x >> 5;
  const int c = lane & 15, g = lane >> 4;
  const int z  = blockIdx.z;
  const int m0 = (blockIdx.x * 8 + wave) * 16;
  const int nb = blockIdx.y;                       // == head index
  const h16* wmat = wsw + (size_t)z * (EMB * EMB);
  const h16* arow = valh + (size_t)(m0 + c) * EMB; // this lane's A row

  v8f acc[4] = {};
  for (int k0 = 0; k0 < EMB; k0 += 32) {
    v16h a = load_a_frag(arow, k0, g);
    const int k16 = (k0 >> 4) + g;
#pragma unroll
    for (int t = 0; t < 4; ++t) {
      const int n16 = nb * 4 + t;
      v16h b = load_b_frag(wmat + (((size_t)(n16 * 48 + k16)) << 8) + c * 16);
      acc[t] = wmma_f16(a, b, acc[t]);
    }
  }

  const float* bias = (z == 0) ? bq : (z == 1) ? bk : bv;
  const float bscale = (z == 0) ? 0.125f : 1.0f;
#pragma unroll
  for (int t = 0; t < 4; ++t) {
    const int dd = t * 16 + c;
    const float bj = bias[nb * 64 + dd] * bscale;
#pragma unroll
    for (int r = 0; r < 8; ++r) {
      const int m = m0 + r + 8 * g;          // flat row = s*BATCH + b
      const int s = m >> 1, bat = m & 1;
      const int bh = bat * HEADS + nb;
      const h16 ov = (h16)(acc[t][r] + bj);
      if (z == 0)      qbuf [((size_t)bh * SEQL + s) * HDIM + dd] = ov;
      else if (z == 1) kbuf [((size_t)bh * KROWS + (s + KPAD)) * HDIM + dd] = ov;
      else             vtbuf[((size_t)bh * HDIM + dd) * KROWS + (s + KPAD)] = ov;
    }
  }
}

// ---- Kernel 4: banded flash attention -------------------------------------
// grid (64, 24), 128 threads = 4 waves; wave -> 16 queries of one (b,h).
__global__ __launch_bounds__(128) void band_attn(
    const h16* __restrict__ qbuf, const h16* __restrict__ kbuf,
    const h16* __restrict__ vtbuf, const int* __restrict__ kpm,
    float* __restrict__ out) {
  __shared__ h16 lds_p[4][16 * 32];
  const int lane = threadIdx.x & 31;
  const int wave = threadIdx.x >> 5;
  const int c = lane & 15, g = lane >> 4;
  const int bat = blockIdx.y / HEADS, h = blockIdx.y % HEADS;
  const int bh = bat * HEADS + h;
  const int q0 = (blockIdx.x * 4 + wave) * 16;

  const h16* qrow = qbuf + ((size_t)bh * SEQL + q0 + c) * HDIM;
  v16h aq0 = load_a_frag(qrow, 0, g);
  v16h aq1 = load_a_frag(qrow, 32, g);

  float mrun[8], lrun[8], alpha[8];
  v8f o[4] = {};
#pragma unroll
  for (int r = 0; r < 8; ++r) { mrun[r] = NEGINF; lrun[r] = 0.f; }

  const h16* kbase = kbuf  + (size_t)bh * KROWS * HDIM;
  const h16* vbase = vtbuf + (size_t)bh * HDIM * KROWS;
  h16* myp = lds_p[wave];

  for (int it = 0; it < 17; ++it) {            // keys [q0-256, q0+288)
    const int kb = q0 - 256 + 32 * it;
    v8f sc[2];
#pragma unroll
    for (int sub = 0; sub < 2; ++sub) {
      const int key = kb + sub * 16 + c;       // this lane's key column
      const h16* krow = kbase + (size_t)(key + KPAD) * HDIM;
      v16h b0 = load_b_frag(krow + 16 * g);        // B[d][key]=K[key][d]
      v16h b1 = load_b_frag(krow + 32 + 16 * g);
      v8f s = {};
      s = wmma_f16(aq0, b0, s);
      s = wmma_f16(aq1, b1, s);
      const bool inr = (key >= 0) && (key < SEQL);
      const float madd = (inr && kpm[bat * SEQL + key] != 0) ? MASKNEG : 0.f;
#pragma unroll
      for (int r = 0; r < 8; ++r) {
        const int qi = q0 + r + 8 * g;
        const bool valid = inr && (key >= qi - 256) && (key <= qi + 256);
        s[r] = valid ? (s[r] + madd) : NEGINF;
      }
      sc[sub] = s;
    }
    // online softmax: 16-lane reductions match C-tile row layout
#pragma unroll
    for (int r = 0; r < 8; ++r) {
      float rmax = fmaxf(sc[0][r], sc[1][r]);
      rmax = fmaxf(rmax, __shfl_xor(rmax, 1, 16));
      rmax = fmaxf(rmax, __shfl_xor(rmax, 2, 16));
      rmax = fmaxf(rmax, __shfl_xor(rmax, 4, 16));
      rmax = fmaxf(rmax, __shfl_xor(rmax, 8, 16));
      const float mo = mrun[r];
      const float mn = fmaxf(mo, rmax);
      const bool dead = (mn == NEGINF);
      const float a = dead ? 1.f : __expf(mo - mn);
      const float p0 = dead ? 0.f : __expf(sc[0][r] - mn);
      const float p1 = dead ? 0.f : __expf(sc[1][r] - mn);
      sc[0][r] = p0; sc[1][r] = p1;
      float rs = p0 + p1;
      rs += __shfl_xor(rs, 1, 16);
      rs += __shfl_xor(rs, 2, 16);
      rs += __shfl_xor(rs, 4, 16);
      rs += __shfl_xor(rs, 8, 16);
      mrun[r] = mn;
      lrun[r] = lrun[r] * a + rs;
      alpha[r] = a;
    }
#pragma unroll
    for (int t = 0; t < 4; ++t)
#pragma unroll
      for (int r = 0; r < 8; ++r) o[t][r] *= alpha[r];

    // P: C layout -> row-major 16x32 in LDS, reload as A fragment
#pragma unroll
    for (int sub = 0; sub < 2; ++sub)
#pragma unroll
      for (int r = 0; r < 8; ++r)
        myp[(r + 8 * g) * 32 + sub * 16 + c] = (h16)sc[sub][r];
    asm volatile("s_wait_dscnt 0x0" ::: "memory");
    v16h ap = load_a_frag(myp + c * 32, 0, g);

#pragma unroll
    for (int t = 0; t < 4; ++t) {               // B[key][dd] = Vt[dd][key]
      const h16* vrow = vbase + (size_t)(t * 16 + c) * KROWS
                        + (kb + KPAD) + 16 * g;
      v16h bvf = load_b_frag(vrow);
      o[t] = wmma_f16(ap, bvf, o[t]);
    }
  }

#pragma unroll
  for (int t = 0; t < 4; ++t)
#pragma unroll
    for (int r = 0; r < 8; ++r) {
      const int srow = q0 + r + 8 * g;
      out[((size_t)srow * BATCH + bat) * EMB + h * HDIM + t * 16 + c] =
          o[t][r] / lrun[r];
    }
}

// ---- host glue ------------------------------------------------------------
extern "C" void kernel_launch(void* const* d_in, const int* in_sizes, int n_in,
                              void* d_out, int out_size, void* d_ws, size_t ws_size,
                              hipStream_t stream) {
  (void)in_sizes; (void)n_in; (void)out_size; (void)ws_size;
  const float* val = (const float*)d_in[0];
  const int*   kpm = (const int*)d_in[1];
  const float* Wq  = (const float*)d_in[2];
  const float* bq  = (const float*)d_in[3];
  const float* Wk  = (const float*)d_in[4];
  const float* bk  = (const float*)d_in[5];
  const float* Wv  = (const float*)d_in[6];
  const float* bv  = (const float*)d_in[7];
  float* out = (float*)d_out;

  char* ws = (char*)d_ws;                      // ~57.3 MB total
  h16* valh  = (h16*)(ws);                     // 12,582,912 B
  h16* wsw   = (h16*)(ws + 12582912);          //  3,538,944 B
  h16* qbuf  = (h16*)(ws + 16121856);          // 12,582,912 B
  h16* kbuf  = (h16*)(ws + 28704768);          // 14,254,080 B
  h16* vtbuf = (h16*)(ws + 42958848);          // 14,254,080 B

  conv_val<<<(MROWS * EMB) / 256, 256, 0, stream>>>(val, valh);
  conv_w<<<(3 * EMB * EMB) / 256, 256, 0, stream>>>(Wq, Wk, Wv, wsw);
  qkv_gemm<<<dim3(64, 12, 3), 256, 0, stream>>>(valh, wsw, bq, bk, bv,
                                                qbuf, kbuf, vtbuf);
  band_attn<<<dim3(64, BATCH * HEADS), 128, 0, stream>>>(qbuf, kbuf, vtbuf,
                                                         kpm, out);
}
